// MultiHeadedAttention_27144193311447
// MI455X (gfx1250) — matrix operand order repaired and checked by reading, not compiled, not run on hardware
//
#include <hip/hip_runtime.h>
#include <hip/hip_bf16.h>

// ---------------------------------------------------------------------------
// MI455X (gfx1250) fused multi-head attention, f16 WMMA path.
//   B=8 S=2048 D=512 H=8 DK=64 WD=64
// Phase 1: QKV projection GEMMs (f32 in -> f16 out; V stored transposed)
// Phase 2: flash attention; K/V tiles double-buffered in LDS via async
//          global->LDS copies; bias MLP in packed f16 (VOP3P); WMMA everywhere
// Phase 3: output projection GEMM (f16 in -> f32 out)
// Workspace: Qh/Kh/Vt/X2, 16MB each = 64MB of d_ws.
// ---------------------------------------------------------------------------

typedef __attribute__((ext_vector_type(16))) _Float16 v16h;
typedef __attribute__((ext_vector_type(8)))  _Float16 v8h;
typedef __attribute__((ext_vector_type(2)))  _Float16 h2;
typedef __attribute__((ext_vector_type(8)))  float    v8f;
typedef __attribute__((ext_vector_type(4)))  int      v4i;

#define WMMA_F16(a, b, c) \
  __builtin_amdgcn_wmma_f32_16x16x32_f16(false, (a), false, (b), (short)0, (c), false, false)

constexpr int Bb = 8, Ss = 2048, Dd = 512, Hh = 8, DK = 64, WD = 64;

// ---- gfx1250 async global->LDS (ASYNCcnt) with portable fallback ----------
#if defined(__has_builtin)
#if __has_builtin(__builtin_amdgcn_global_load_async_to_lds_b128)
#define HAVE_ASYNC 1
#endif
#endif
#ifndef HAVE_ASYNC
#define HAVE_ASYNC 0
#endif

__device__ __forceinline__ void cp16_to_lds(_Float16* dst, const _Float16* src) {
#if HAVE_ASYNC
  __builtin_amdgcn_global_load_async_to_lds_b128(
      (__attribute__((address_space(1))) v4i*)(src),
      (__attribute__((address_space(3))) v4i*)(dst), 0, 0);
#else
  *(v8h*)dst = *(const v8h*)src;
#endif
}

__device__ __forceinline__ void async_fence() {
#if HAVE_ASYNC
#if __has_builtin(__builtin_amdgcn_s_wait_asynccnt)
  __builtin_amdgcn_s_wait_asynccnt(0);
#else
  asm volatile("s_wait_asynccnt 0" ::: "memory");
#endif
#endif
}

__device__ __forceinline__ v16h mk16(v8h lo, v8h hi) {
  return __builtin_shufflevector(lo, hi, 0, 1, 2, 3, 4, 5, 6, 7,
                                 8, 9, 10, 11, 12, 13, 14, 15);
}

// ---------------------------------------------------------------------------
// Phase 1: Q/K/V projection. grid = (M/128, N/16, 3), block = 256 (8 waves).
// ---------------------------------------------------------------------------
__global__ __launch_bounds__(256)
void k_qkv_proj(const float* __restrict__ q_in, const float* __restrict__ k_in,
                const float* __restrict__ v_in,
                const float* __restrict__ Wq, const float* __restrict__ bq,
                const float* __restrict__ Wk, const float* __restrict__ bk,
                const float* __restrict__ Wv, const float* __restrict__ bv,
                _Float16* __restrict__ Qh, _Float16* __restrict__ Kh,
                _Float16* __restrict__ Vt) {
  const int mode = blockIdx.z;
  const float* X    = (mode == 0) ? q_in : (mode == 1) ? k_in : v_in;
  const float* W    = (mode == 0) ? Wq   : (mode == 1) ? Wk   : Wv;
  const float* bias = (mode == 0) ? bq   : (mode == 1) ? bk   : bv;
  _Float16*    out  = (mode == 0) ? Qh   : (mode == 1) ? Kh   : Vt;

  const int lane = threadIdx.x & 31;
  const int wid  = threadIdx.x >> 5;
  const int half = lane >> 4;
  const int sub  = lane & 15;

  const int M0 = blockIdx.x * 128 + wid * 16;
  const int N0 = blockIdx.y * 16;

  const float* xrow = X + (size_t)(M0 + sub) * Dd;   // A row: M = lane&15
  const float* wrow = W + (size_t)(N0 + sub) * Dd;   // B col: N = lane&15

  v8f c = {};
  for (int ks = 0; ks < Dd; ks += 32) {
    v16h a, b;
    const float* pa = xrow + ks + half * 16;         // A: K = half*16+{0..15}
#pragma unroll
    for (int i = 0; i < 8; ++i) { a[i] = (_Float16)pa[i]; a[8 + i] = (_Float16)pa[8 + i]; }
    const float* pb = wrow + ks + half * 16;         // B: K = half*16+{0..15}
#pragma unroll
    for (int i = 0; i < 16; ++i) b[i] = (_Float16)pb[i];
    c = WMMA_F16(a, b, c);
  }

  const float bv_ = bias[N0 + sub];
#pragma unroll
  for (int r = 0; r < 8; ++r) {                      // C: M = r+8*half, N = lane&15
    const int m  = M0 + r + 8 * half;
    const int o  = N0 + sub;
    const int bb = m >> 11, s = m & 2047;
    const int hh = o >> 6,  dk = o & 63;
    const float val = c[r] + bv_;
    if (mode < 2)
      out[(((size_t)(bb * Hh + hh)) * Ss + s) * DK + dk] = (_Float16)val;   // [B,H,S,DK]
    else
      out[(((size_t)(bb * Hh + hh)) * DK + dk) * Ss + s] = (_Float16)val;   // [B,H,DK,S]
  }
}

// ---------------------------------------------------------------------------
// Phase 2: flash attention. grid = (S/128, B*H), block = 256 (8 waves).
// Each wave owns 16 query rows; K/V tiles (64 keys) are staged per-block in
// LDS with double buffering (async global->LDS), shared by all 8 waves.
// ---------------------------------------------------------------------------
__global__ __launch_bounds__(256)
void k_flash(const _Float16* __restrict__ Qh, const _Float16* __restrict__ Kh,
             const _Float16* __restrict__ Vt,
             const int* __restrict__ mask, const float* __restrict__ segw,
             const float* __restrict__ Wl, const float* __restrict__ bl,
             const float* __restrict__ Wr, const float* __restrict__ br,
             _Float16* __restrict__ X2) {
  // rows padded to 72 halves (36 dwords): 16-row fragment reads are
  // bank-conflict-free (gcd(36,64)=4 -> 16 distinct 4-bank groups).
  __shared__ _Float16 s_k[2][64][72];
  __shared__ _Float16 s_v[2][64][72];
  __shared__ _Float16 s_p[8][16][72];
  __shared__ h2 s_wl2[32], s_bl2[32], s_wr2[32];

  const int t = threadIdx.x;
  if (t < 32) {                                      // pack MLP weights as f16x2
    h2 a, b, c;
    a[0] = (_Float16)Wl[2 * t]; a[1] = (_Float16)Wl[2 * t + 1];
    b[0] = (_Float16)bl[2 * t]; b[1] = (_Float16)bl[2 * t + 1];
    c[0] = (_Float16)Wr[2 * t]; c[1] = (_Float16)Wr[2 * t + 1];
    s_wl2[t] = a; s_bl2[t] = b; s_wr2[t] = c;
  }

  const int lane = t & 31, wid = t >> 5;
  const int half = lane >> 4, sub = lane & 15;
  const int bh = blockIdx.y, b = bh >> 3, h = bh & 7;
  const int q0 = blockIdx.x * 128 + wid * 16;

  const _Float16* Qbase = Qh + (size_t)bh * Ss * DK;
  const _Float16* Kbase = Kh + (size_t)bh * Ss * DK;
  const _Float16* Vbase = Vt + (size_t)bh * DK * Ss;
  const float*    swb   = segw + (size_t)b * Ss * Ss;
  const float     br0   = br[0];

  // stage one 64-key K tile + V tile into LDS buffer `buf` (2x16B per thread)
  auto stage = [&](int buf, int kb) {
#pragma unroll
    for (int i = 0; i < 2; ++i) {
      const int chunk = t * 2 + i;                   // 0..511
      const int row = chunk >> 3, c8 = chunk & 7;
      cp16_to_lds(&s_k[buf][row][c8 * 8],
                  Kbase + (size_t)(kb + row) * DK + c8 * 8);
      cp16_to_lds(&s_v[buf][row][c8 * 8],
                  Vbase + (size_t)row * Ss + kb + c8 * 8);
    }
  };

  // Q A-fragments held in VGPRs for the whole kernel (two 16x32 fragments)
  v16h aq[2];
  {
    const _Float16* qrow = Qbase + (size_t)(q0 + sub) * DK;
#pragma unroll
    for (int s2 = 0; s2 < 2; ++s2) {
      const _Float16* p = qrow + s2 * 32 + half * 8;
      aq[s2] = mk16(*(const v8h*)p, *(const v8h*)(p + 16));
    }
  }

  float mrow[8], lrow[8], msk[8];
#pragma unroll
  for (int r = 0; r < 8; ++r) {
    mrow[r] = -3.0e38f;
    lrow[r] = 0.0f;
    msk[r]  = (float)mask[b * Ss + q0 + r + 8 * half];
  }
  v8f z = {};
  v8f o[4] = {z, z, z, z};

  stage(0, 0);
  async_fence();
  __syncthreads();                                   // buffer 0 + MLP weights ready

  const h2 c06 = { (_Float16)0.6f, (_Float16)0.6f };
  const h2 c04 = { (_Float16)0.4f, (_Float16)0.4f };

  for (int kb = 0; kb < Ss; kb += 64) {
    const int cur = (kb >> 6) & 1;
    if (kb + 64 < Ss) stage(cur ^ 1, kb + 64);       // overlap copy with compute
    __builtin_prefetch(swb + (size_t)(q0 + sub) * Ss + kb + 64, 0, 1);

    // ---- scores: S = Q K^T (8 WMMAs), B-fragments from LDS ----
    v8f c[4] = {z, z, z, z};
#pragma unroll
    for (int nt = 0; nt < 4; ++nt) {
#pragma unroll
      for (int s2 = 0; s2 < 2; ++s2) {
        const _Float16* kr = &s_k[cur][nt * 16 + sub][s2 * 32 + half * 8];
        v16h bk_ = mk16(*(const v8h*)kr, *(const v8h*)(kr + 16));
        c[nt] = WMMA_F16(aq[s2], bk_, c[nt]);
      }
    }

    // ---- fused segment-bias MLP (packed f16) + scale + query-row mask ----
#pragma unroll
    for (int nt = 0; nt < 4; ++nt) {
#pragma unroll
      for (int r = 0; r < 8; ++r) {
        const int q   = q0 + r + 8 * half;
        const int key = kb + nt * 16 + sub;
        const float u  = swb[(size_t)q * Ss + key];
        const float sw = 1.0f / __logf(2.71828182845904523f + u);
        const _Float16 swh = (_Float16)sw;
        h2 sw2; sw2[0] = swh; sw2[1] = swh;
        h2 acc2 = { (_Float16)0.0f, (_Float16)0.0f };
#pragma unroll 8
        for (int j = 0; j < WD / 2; ++j) {
          h2 tt = sw2 * s_wl2[j] + s_bl2[j];         // v_pk_fma_f16
          h2 ab = __builtin_elementwise_abs(tt);
          tt = tt * c06 + ab * c04;                  // leaky_relu(0.2) = 0.6t+0.4|t|
          acc2 = acc2 + tt * s_wr2[j];               // v_pk_fma_f16
        }
        const float bias = (float)acc2[0] + (float)acc2[1] + br0;
        c[nt][r] = (msk[r] != 0.0f) ? (c[nt][r] * 0.125f + bias) : -1.0e9f;
      }
    }

    // ---- online softmax (row reductions across the 16-lane row group) ----
#pragma unroll
    for (int r = 0; r < 8; ++r) {
      float mx = fmaxf(fmaxf(c[0][r], c[1][r]), fmaxf(c[2][r], c[3][r]));
#pragma unroll
      for (int i = 1; i < 16; i <<= 1) mx = fmaxf(mx, __shfl_xor(mx, i, 32));
      const float mnew = fmaxf(mrow[r], mx);
      const float corr = __expf(mrow[r] - mnew);
      float rsum = 0.0f;
#pragma unroll
      for (int nt = 0; nt < 4; ++nt) {
        const float p = __expf(c[nt][r] - mnew);
        c[nt][r] = p;
        rsum += p;
      }
#pragma unroll
      for (int i = 1; i < 16; i <<= 1) rsum += __shfl_xor(rsum, i, 32);
      lrow[r] = lrow[r] * corr + rsum;
      mrow[r] = mnew;
#pragma unroll
      for (int dt = 0; dt < 4; ++dt) o[dt][r] *= corr;
    }

    // ---- P: C-layout -> A-layout via per-wave LDS (same-wave DS in-order) ----
#pragma unroll
    for (int nt = 0; nt < 4; ++nt)
#pragma unroll
      for (int r = 0; r < 8; ++r)
        s_p[wid][r + 8 * half][nt * 16 + sub] = (_Float16)c[nt][r];

    v16h ap[2];
#pragma unroll
    for (int s2 = 0; s2 < 2; ++s2) {
      const _Float16* p = &s_p[wid][sub][s2 * 32 + half * 8];
      ap[s2] = mk16(*(const v8h*)p, *(const v8h*)(p + 16));
    }

    // ---- O += P V (8 WMMAs), B-fragments from LDS (V pre-transposed) ----
#pragma unroll
    for (int dt = 0; dt < 4; ++dt) {
#pragma unroll
      for (int s2 = 0; s2 < 2; ++s2) {
        const _Float16* vr = &s_v[cur][dt * 16 + sub][s2 * 32 + half * 8];
        v16h bv_ = mk16(*(const v8h*)vr, *(const v8h*)(vr + 16));
        o[dt] = WMMA_F16(ap[s2], bv_, o[dt]);
      }
    }

    async_fence();                                   // staging writes landed
    __syncthreads();                                 // all waves done with `cur`
  }

  // ---- epilogue: O / l -> X2 [B,S,H*DK] f16 ----
#pragma unroll
  for (int dt = 0; dt < 4; ++dt) {
#pragma unroll
    for (int r = 0; r < 8; ++r) {
      const int q  = q0 + r + 8 * half;
      const int dk = dt * 16 + sub;
      X2[((size_t)(b * Ss + q)) * Dd + h * DK + dk] = (_Float16)(o[dt][r] / lrow[r]);
    }
  }
}

// ---------------------------------------------------------------------------
// Phase 3: output projection. grid = (M/128, N/16), block = 256.
// ---------------------------------------------------------------------------
__global__ __launch_bounds__(256)
void k_outproj(const _Float16* __restrict__ X2, const float* __restrict__ Wo,
               const float* __restrict__ bo, float* __restrict__ out) {
  const int lane = threadIdx.x & 31;
  const int wid  = threadIdx.x >> 5;
  const int half = lane >> 4;
  const int sub  = lane & 15;

  const int M0 = blockIdx.x * 128 + wid * 16;
  const int N0 = blockIdx.y * 16;

  const _Float16* xrow = X2 + (size_t)(M0 + sub) * Dd;
  const float*    wrow = Wo + (size_t)(N0 + sub) * Dd;

  v8f c = {};
  for (int ks = 0; ks < Dd; ks += 32) {
    const _Float16* pa = xrow + ks + half * 16;
    v16h a = mk16(*(const v8h*)pa, *(const v8h*)(pa + 8));
    v16h b;
    const float* pb = wrow + ks + half * 16;
#pragma unroll
    for (int i = 0; i < 16; ++i) b[i] = (_Float16)pb[i];
    c = WMMA_F16(a, b, c);
  }

  const float bo_ = bo[N0 + sub];
#pragma unroll
  for (int r = 0; r < 8; ++r)
    out[(size_t)(M0 + r + 8 * half) * Dd + (N0 + sub)] = c[r] + bo_;
}

// ---------------------------------------------------------------------------
extern "C" void kernel_launch(void* const* d_in, const int* in_sizes, int n_in,
                              void* d_out, int out_size, void* d_ws, size_t ws_size,
                              hipStream_t stream) {
  const float* query = (const float*)d_in[0];
  const float* key   = (const float*)d_in[1];
  const float* value = (const float*)d_in[2];
  const int*   mask  = (const int*)  d_in[3];
  const float* segw  = (const float*)d_in[4];
  const float* Wq = (const float*)d_in[5],  *bq = (const float*)d_in[6];
  const float* Wk = (const float*)d_in[7],  *bk = (const float*)d_in[8];
  const float* Wv = (const float*)d_in[9],  *bv = (const float*)d_in[10];
  const float* Wo = (const float*)d_in[11], *bo = (const float*)d_in[12];
  const float* Wl = (const float*)d_in[13], *bl = (const float*)d_in[14];
  const float* Wr = (const float*)d_in[15], *br = (const float*)d_in[16];

  const size_t nqkv = (size_t)Bb * Hh * Ss * DK;     // 8,388,608 halves per buffer
  _Float16* Qh = (_Float16*)d_ws;
  _Float16* Kh = Qh + nqkv;
  _Float16* Vt = Kh + nqkv;
  _Float16* X2 = Vt + nqkv;                          // 64MB total in d_ws

  dim3 blk(256);
  k_qkv_proj<<<dim3(128, 32, 3), blk, 0, stream>>>(query, key, value,
                                                   Wq, bq, Wk, bk, Wv, bv,
                                                   Qh, Kh, Vt);
  k_flash<<<dim3(Ss / 128, Bb * Hh), blk, 0, stream>>>(Qh, Kh, Vt, mask, segw,
                                                       Wl, bl, Wr, br, X2);
  k_outproj<<<dim3(128, 32), blk, 0, stream>>>(X2, Wo, bo, (float*)d_out);
}
